// MultiMetricPredictor_34033320853531
// MI455X (gfx1250) — matrix operand order repaired and checked by reading, not compiled, not run on hardware
//
#include <hip/hip_runtime.h>

// ---------------------------------------------------------------------------
// CDNA5 (gfx1250) implementation: transformer encoder + GRU decoder.
// All GEMMs run through v_wmma_f32_16x16x32_bf16 (wave32, 16x16 tiles).
// Each GEMM wave computes a 16x64 output strip (4 accumulators) so the
// A fragment is loaded once per k-chunk and reused across 4 WMMAs.
// ---------------------------------------------------------------------------

typedef __attribute__((ext_vector_type(16))) __bf16 v16bf;
typedef __attribute__((ext_vector_type(8)))  float  v8f;

#define TT 120      // lookback
#define DD 128      // model dim
#define NB 1024     // batch
#define HOR_ 90
#define GM 32       // GRU rows per block

// ---------------- scalar helpers -------------------------------------------

__device__ __forceinline__ float bf2f(__bf16 h) {
    unsigned short s = __builtin_bit_cast(unsigned short, h);
    unsigned u = ((unsigned)s) << 16;
    return __builtin_bit_cast(float, u);
}
__device__ __forceinline__ __bf16 f2bf(float f) {
    unsigned u = __builtin_bit_cast(unsigned, f);
    unsigned r = u + 0x7FFFu + ((u >> 16) & 1u);   // round to nearest even
    unsigned short s = (unsigned short)(r >> 16);
    return __builtin_bit_cast(__bf16, s);
}

// ---------------- WMMA fragment helpers ------------------------------------
// A (16x32 bf16, row-major [M,K]): lane L (m = L%16, half = L/16) holds
//   elems 0..7  -> k = half*8 + e        (16B contiguous)
//   elems 8..15 -> k = 16 + half*8 + e   (16B contiguous)
__device__ __forceinline__ v16bf frag_a(const __bf16* rowBase, int half) {
    v16bf a;
    ((uint4*)&a)[0] = *(const uint4*)(rowBase + half * 8);
    ((uint4*)&a)[1] = *(const uint4*)(rowBase + 16 + half * 8);
    return a;
}
// B (32x16 bf16) with weights stored native [N,K] row-major: lane L
// (n = L%16, half = L/16) holds elems 0..7 -> k = half*8 + e,
// elems 8..15 -> k = 16 + half*8 + e (same order as A).
__device__ __forceinline__ v16bf frag_b(const __bf16* rowBase, int half) {
    v16bf b;
    ((uint4*)&b)[0] = *(const uint4*)(rowBase + half * 8);
    ((uint4*)&b)[1] = *(const uint4*)(rowBase + 16 + half * 8);
    return b;
}
__device__ __forceinline__ v16bf frag_zero() {
    v16bf z;
    uint4 q; q.x = q.y = q.z = q.w = 0u;
    ((uint4*)&z)[0] = q;
    ((uint4*)&z)[1] = q;
    return z;
}
__device__ __forceinline__ v8f acc_zero() {
    v8f a = {0.f, 0.f, 0.f, 0.f, 0.f, 0.f, 0.f, 0.f};
    return a;
}
__device__ __forceinline__ v8f wmma_bf16(v16bf a, v16bf b, v8f c) {
    return __builtin_amdgcn_wmma_f32_16x16x32_bf16(false, a, false, b, (short)0, c, false, false);
}

// ---------------- weight convert f32 -> bf16 with K padding ----------------

__global__ void cvt_kernel(const float* __restrict__ src, __bf16* __restrict__ dst,
                           int N, int K, int Kpad, int stride, int soff) {
    int i = blockIdx.x * blockDim.x + threadIdx.x;
    if (i >= N * Kpad) return;
    int n = i / Kpad, k = i % Kpad;
    float v = (k < K) ? src[(size_t)n * stride + soff + k] : 0.f;
    dst[i] = f2bf(v);
}

// ---------------- build concat([x, sym_emb]) input as bf16 [B*T, 64] -------

__global__ void build_in_kernel(const float* __restrict__ x, const int* __restrict__ sym_id,
                                const float* __restrict__ sym_emb, __bf16* __restrict__ A0,
                                int total) {
    int i = blockIdx.x * blockDim.x + threadIdx.x;
    if (i >= total) return;
    int c = i & 63;
    int row = i >> 6;
    int b = row / TT;
    float v;
    if (c < 32)      v = x[(size_t)row * 32 + c];
    else if (c < 48) v = sym_emb[(size_t)sym_id[b] * 16 + (c - 32)];
    else             v = 0.f;
    A0[i] = f2bf(v);
}

// ---------------- add sinusoidal positional encoding -----------------------

__global__ void add_pe_kernel(float* __restrict__ h, int total) {
    int i = blockIdx.x * blockDim.x + threadIdx.x;
    if (i >= total) return;
    int d = i & 127;
    int t = (i >> 7) % TT;
    int pi = d >> 1;
    float div = expf((float)(2 * pi) * (-9.210340371976184f / 128.0f));
    float ang = (float)t * div;
    h[i] += (d & 1) ? cosf(ang) : sinf(ang);
}

// ---------------- LayerNorm: one wave per row of 128, writes bf16 ----------

__global__ void __launch_bounds__(256) ln_kernel(const float* __restrict__ X,
                                                 const float* __restrict__ w,
                                                 const float* __restrict__ b,
                                                 __bf16* __restrict__ Y, int rows) {
    int wave = (blockIdx.x * blockDim.x + threadIdx.x) >> 5;
    if (wave >= rows) return;
    int lane = threadIdx.x & 31;
    const float* xr = X + (size_t)wave * DD;
    float4 v = *(const float4*)(xr + lane * 4);
    float s = v.x + v.y + v.z + v.w;
    float q = v.x * v.x + v.y * v.y + v.z * v.z + v.w * v.w;
    #pragma unroll
    for (int off = 16; off; off >>= 1) {
        s += __shfl_xor(s, off, 32);
        q += __shfl_xor(q, off, 32);
    }
    float mean = s * (1.f / 128.f);
    float var  = q * (1.f / 128.f) - mean * mean;
    float inv  = rsqrtf(var + 1e-5f);
    int d0 = lane * 4;
    __bf16* yr = Y + (size_t)wave * DD;
    float vv[4] = {v.x, v.y, v.z, v.w};
    #pragma unroll
    for (int j = 0; j < 4; ++j)
        yr[d0 + j] = f2bf((vv[j] - mean) * inv * w[d0 + j] + b[d0 + j]);
}

// ---------------- generic strip WMMA GEMM ----------------------------------
// C[M,N] = A[M,K](bf16) * W[N,K]^T(bf16) + bias; optional relu / residual /
// f32 and bf16 outputs. K multiple of 32, M multiple of 16, N multiple of 64.
// Each wave computes a 16x64 strip: A fragment reused across 4 WMMAs.

__global__ void __launch_bounds__(256) gemm_bf16_kernel(
    const __bf16* __restrict__ A, const __bf16* __restrict__ W,
    const float* __restrict__ bias, const float* __restrict__ resid,
    float* __restrict__ Cf, __bf16* __restrict__ Cb,
    int M, int N, int K, int relu) {
    int wave = (blockIdx.x * blockDim.x + threadIdx.x) >> 5;
    int stripsN = N >> 6;
    int totalStrips = (M >> 4) * stripsN;
    if (wave >= totalStrips) return;
    int tm = wave / stripsN, ts = wave % stripsN;
    int lane = threadIdx.x & 31;
    int m16 = lane & 15, half = lane >> 4;
    const __bf16* aBase = A + (size_t)(tm * 16 + m16) * K;
    const __bf16* wB = W + (size_t)(ts * 64 + m16) * K;   // sub-tile j: + j*16*K
    v8f acc[4];
    #pragma unroll
    for (int j = 0; j < 4; ++j) acc[j] = acc_zero();
    for (int kk = 0; kk < K; kk += 32) {
        v16bf a = frag_a(aBase + kk, half);
        #pragma unroll
        for (int j = 0; j < 4; ++j)
            acc[j] = wmma_bf16(a, frag_b(wB + (size_t)j * 16 * K + kk, half), acc[j]);
    }
    #pragma unroll
    for (int j = 0; j < 4; ++j) {
        int n = ts * 64 + j * 16 + m16;
        float bv = bias ? bias[n] : 0.f;
        #pragma unroll
        for (int v = 0; v < 8; ++v) {
            int m = tm * 16 + v + 8 * half;
            size_t idx = (size_t)m * N + n;
            float val = acc[j][v] + bv;
            if (relu) val = fmaxf(val, 0.f);
            if (resid) val += resid[idx];
            if (Cf) Cf[idx] = val;
            if (Cb) Cb[idx] = f2bf(val);
        }
    }
}

// ---------------- multi-head attention: one block per (b, head) ------------
// qkv bf16 [B*T, 384] = [q(128) | k(128) | v(128)]; T padded 120 -> 128.

__global__ void __launch_bounds__(256) attn_kernel(const __bf16* __restrict__ qkv,
                                                   __bf16* __restrict__ attn_out) {
    int bh = blockIdx.x;
    int b  = bh >> 2;
    int hh = bh & 3;
    int tid  = threadIdx.x;
    int wave = tid >> 5;
    int lane = tid & 31;
    int m16 = lane & 15, half = lane >> 4;

    __shared__ __bf16 S[128 * 128];   // scores, then probabilities (32 KB)
    __shared__ __bf16 Vt[32 * 128];   // V transposed [c][t'] (8 KB)

    // stage V transposed
    for (int i = tid; i < 32 * 128; i += 256) {
        int c = i >> 7, t = i & 127;
        Vt[i] = (t < TT) ? qkv[((size_t)(b * TT + t)) * 384 + 256 + hh * 32 + c]
                         : f2bf(0.f);
    }

    // phase 1: S = Q K^T / sqrt(32).  wave -> tile row, loop over tile cols.
    {
        int tq = wave * 16 + m16;  // A row (query time)
        v16bf a = (tq < TT)
            ? frag_a(qkv + ((size_t)(b * TT + tq)) * 384 + hh * 32, half)
            : frag_zero();
        for (int tn = 0; tn < 8; ++tn) {
            int tk = tn * 16 + m16;  // B col (key time)
            v16bf bf_ = (tk < TT)
                ? frag_b(qkv + ((size_t)(b * TT + tk)) * 384 + 128 + hh * 32, half)
                : frag_zero();
            v8f acc = acc_zero();
            acc = wmma_bf16(a, bf_, acc);
            #pragma unroll
            for (int v = 0; v < 8; ++v) {
                int rrow = wave * 16 + v + 8 * half;
                S[rrow * 128 + tn * 16 + m16] = f2bf(acc[v] * 0.17677669529663687f);
            }
        }
    }
    __syncthreads();

    // softmax over key dim (120 valid), thread per row, in place bf16
    if (tid < 128) {
        int r = tid;
        if (r < TT) {
            float mx = -3.0e38f;
            for (int c = 0; c < TT; ++c) mx = fmaxf(mx, bf2f(S[r * 128 + c]));
            float sum = 0.f;
            for (int c = 0; c < TT; ++c) sum += expf(bf2f(S[r * 128 + c]) - mx);
            float inv = 1.f / sum;
            for (int c = 0; c < TT; ++c)
                S[r * 128 + c] = f2bf(expf(bf2f(S[r * 128 + c]) - mx) * inv);
            for (int c = TT; c < 128; ++c) S[r * 128 + c] = f2bf(0.f);
        } else {
            for (int c = 0; c < 128; ++c) S[r * 128 + c] = f2bf(0.f);
        }
    }
    __syncthreads();

    // phase 2: O = P @ V.  wave -> M tile; A fragment reused for both N tiles.
    {
        int tm = wave;
        v8f acc0 = acc_zero(), acc1 = acc_zero();
        for (int kk = 0; kk < 128; kk += 32) {
            v16bf a = frag_a(&S[(tm * 16 + m16) * 128 + kk], half);
            acc0 = wmma_bf16(a, frag_b(&Vt[(m16) * 128 + kk], half), acc0);
            acc1 = wmma_bf16(a, frag_b(&Vt[(16 + m16) * 128 + kk], half), acc1);
        }
        #pragma unroll
        for (int v = 0; v < 8; ++v) {
            int t = tm * 16 + v + 8 * half;
            if (t < TT) {
                size_t base = ((size_t)(b * TT + t)) * 128 + hh * 32;
                attn_out[base + m16]      = f2bf(acc0[v]);
                attn_out[base + 16 + m16] = f2bf(acc1[v]);
            }
        }
    }
}

// ---------------- attention pooling + ctx-input build ----------------------

__global__ void __launch_bounds__(128) pool_kernel(
    const float* __restrict__ h, const float* __restrict__ pool_w,
    const float* __restrict__ pool_b, const int* __restrict__ sym_id,
    const float* __restrict__ sym_emb, const int* __restrict__ regime_id,
    const float* __restrict__ reg_emb, __bf16* __restrict__ ctxin) {
    int b = blockIdx.x;
    int tid = threadIdx.x;
    __shared__ float lg[128];
    __shared__ float red[128];
    const float* hb = h + (size_t)b * TT * DD;
    float l = -3.0e38f;
    if (tid < TT) {
        float s = pool_b[0];
        for (int d = 0; d < DD; ++d) s += hb[tid * DD + d] * pool_w[d];
        l = s;
    }
    red[tid] = l;
    __syncthreads();
    for (int off = 64; off; off >>= 1) {
        if (tid < off) red[tid] = fmaxf(red[tid], red[tid + off]);
        __syncthreads();
    }
    float mx = red[0];
    __syncthreads();
    float e = (tid < TT) ? expf(l - mx) : 0.f;
    lg[tid] = e;
    red[tid] = e;
    __syncthreads();
    for (int off = 64; off; off >>= 1) {
        if (tid < off) red[tid] += red[tid + off];
        __syncthreads();
    }
    float inv = 1.f / red[0];
    __syncthreads();
    float p = 0.f;
    for (int t = 0; t < TT; ++t) p += lg[t] * inv * hb[t * DD + tid];
    __bf16* cr = ctxin + (size_t)b * 160;
    cr[tid] = f2bf(p);
    if (tid < 16) cr[128 + tid] = f2bf(sym_emb[(size_t)sym_id[b] * 16 + tid]);
    if (tid < 8)  cr[144 + tid] = f2bf(reg_emb[(size_t)regime_id[b] * 8 + tid]);
    if (tid < 8)  cr[152 + tid] = f2bf(0.f);
}

// ---------------- realized vol: unbiased std over time of feature 0 --------

__global__ void rv_kernel(const float* __restrict__ x, float* __restrict__ rv, int nb) {
    int b = blockIdx.x * blockDim.x + threadIdx.x;
    if (b >= nb) return;
    float s = 0.f;
    for (int t = 0; t < TT; ++t) s += x[((size_t)b * TT + t) * 32];
    float m = s / (float)TT;
    float q = 0.f;
    for (int t = 0; t < TT; ++t) {
        float d = x[((size_t)b * TT + t) * 32] - m;
        q += d * d;
    }
    rv[b] = sqrtf(q / (float)(TT - 1));
}

// ---------------- persistent GRU decoder: 32 batch rows per block ----------

__global__ void __launch_bounds__(256) gru_kernel(
    const float* __restrict__ ctx,        // [B,128] initial hidden
    const float* __restrict__ gi_ctx,     // [B,384] ctx-part of gi (+bih)
    const __bf16* __restrict__ whh_b,     // [384,128]
    const float* __restrict__ gru_bhh,    // [384]
    const float* __restrict__ gru_wih,    // [384,133] (cols 0..4 = pred part)
    const __bf16* __restrict__ mu_w1b,    // [64,128]
    const float* __restrict__ mu_b1,
    const float* __restrict__ mu_w2,      // [5,64]
    const float* __restrict__ mu_b2,
    const __bf16* __restrict__ vol_w1b,
    const float* __restrict__ vol_b1,
    const float* __restrict__ vol_w2,
    const float* __restrict__ vol_b2,
    const float* __restrict__ rv,
    float* __restrict__ out) {            // [B,90,5]
    __shared__ float  hd[GM * 128];
    __shared__ __bf16 hdb[GM * 128];
    __shared__ __bf16 gh[GM * 384];
    __shared__ __bf16 m1[GM * 64];
    __shared__ __bf16 v1[GM * 64];
    __shared__ float  pred[GM * 5];
    __shared__ float  rvs[GM];
    int tid = threadIdx.x;
    int wave = tid >> 5;
    int lane = tid & 31;
    int m16 = lane & 15, half = lane >> 4;
    int b0 = blockIdx.x * GM;

    for (int i = tid; i < GM * 128; i += 256) hd[i] = ctx[(size_t)b0 * 128 + i];
    if (tid < GM * 5) pred[tid] = 0.f;
    if (tid < GM) rvs[tid] = rv[b0 + tid];
    __syncthreads();
    for (int i = tid; i < GM * 128; i += 256) hdb[i] = f2bf(hd[i]);
    __syncthreads();

    for (int step = 0; step < HOR_; ++step) {
        // gh = hd @ whh^T + bhh : 2 (M) x 6 (N strips of 64) = 12 strips
        for (int ss = wave; ss < 12; ss += 8) {
            int tm = ss / 6, sn = ss % 6;
            const __bf16* aBase = &hdb[(tm * 16 + m16) * 128];
            const __bf16* wB = whh_b + (size_t)(sn * 64 + m16) * 128;
            v8f acc[4];
            #pragma unroll
            for (int j = 0; j < 4; ++j) acc[j] = acc_zero();
            #pragma unroll
            for (int kk = 0; kk < 128; kk += 32) {
                v16bf a = frag_a(aBase + kk, half);
                #pragma unroll
                for (int j = 0; j < 4; ++j)
                    acc[j] = wmma_bf16(a, frag_b(wB + (size_t)j * 16 * 128 + kk, half), acc[j]);
            }
            #pragma unroll
            for (int j = 0; j < 4; ++j) {
                int n = sn * 64 + j * 16 + m16;
                float bv = gru_bhh[n];
                #pragma unroll
                for (int v = 0; v < 8; ++v)
                    gh[(tm * 16 + v + 8 * half) * 384 + n] = f2bf(acc[j][v] + bv);
            }
        }
        __syncthreads();

        // gates (VALU): gi = gi_ctx + pred @ wih[:, :5]^T
        for (int i = tid; i < GM * 128; i += 256) {
            int r = i >> 7, d = i & 127;
            int gb = b0 + r;
            float p0 = pred[r * 5 + 0], p1 = pred[r * 5 + 1], p2 = pred[r * 5 + 2];
            float p3 = pred[r * 5 + 3], p4 = pred[r * 5 + 4];
            const float* gic = gi_ctx + (size_t)gb * 384;
            const float* w_r = gru_wih + (size_t)d * 133;
            const float* w_z = gru_wih + (size_t)(128 + d) * 133;
            const float* w_n = gru_wih + (size_t)(256 + d) * 133;
            float ir  = gic[d]       + p0*w_r[0] + p1*w_r[1] + p2*w_r[2] + p3*w_r[3] + p4*w_r[4];
            float iz  = gic[128 + d] + p0*w_z[0] + p1*w_z[1] + p2*w_z[2] + p3*w_z[3] + p4*w_z[4];
            float in2 = gic[256 + d] + p0*w_n[0] + p1*w_n[1] + p2*w_n[2] + p3*w_n[3] + p4*w_n[4];
            float rr = 1.f / (1.f + expf(-(ir + bf2f(gh[r * 384 + d]))));
            float zz = 1.f / (1.f + expf(-(iz + bf2f(gh[r * 384 + 128 + d]))));
            float nn = tanhf(in2 + rr * bf2f(gh[r * 384 + 256 + d]));
            hd[i] = (1.f - zz) * nn + zz * hd[i];
        }
        __syncthreads();
        for (int i = tid; i < GM * 128; i += 256) hdb[i] = f2bf(hd[i]);
        __syncthreads();

        // mu1/vol1 = gelu(hd @ w1^T + b1) : strips = which(2) x tm(2) = 4
        for (int ss = wave; ss < 4; ss += 8) {
            int which = ss >> 1;
            int tm = ss & 1;
            const __bf16* wB = (which ? vol_w1b : mu_w1b) + (size_t)m16 * 128;
            const float*  bb = which ? vol_b1 : mu_b1;
            const __bf16* aBase = &hdb[(tm * 16 + m16) * 128];
            v8f acc[4];
            #pragma unroll
            for (int j = 0; j < 4; ++j) acc[j] = acc_zero();
            #pragma unroll
            for (int kk = 0; kk < 128; kk += 32) {
                v16bf a = frag_a(aBase + kk, half);
                #pragma unroll
                for (int j = 0; j < 4; ++j)
                    acc[j] = wmma_bf16(a, frag_b(wB + (size_t)j * 16 * 128 + kk, half), acc[j]);
            }
            __bf16* dst = which ? v1 : m1;
            #pragma unroll
            for (int j = 0; j < 4; ++j) {
                int n = j * 16 + m16;
                float bv = bb[n];
                #pragma unroll
                for (int v = 0; v < 8; ++v) {
                    float tvl = acc[j][v] + bv;
                    float g = 0.5f * tvl * (1.f + erff(tvl * 0.7071067811865476f));
                    dst[(tm * 16 + v + 8 * half) * 64 + n] = f2bf(g);
                }
            }
        }
        __syncthreads();

        // heads: 64 -> 5 (VALU), tanh / softplus, write preds
        if (tid < GM * 5) {
            int r = tid / 5, mm = tid % 5;
            float am = mu_b2[mm], av = vol_b2[mm];
            const float* wm = mu_w2 + mm * 64;
            const float* wv = vol_w2 + mm * 64;
            for (int j = 0; j < 64; ++j) {
                am += bf2f(m1[r * 64 + j]) * wm[j];
                av += bf2f(v1[r * 64 + j]) * wv[j];
            }
            float mu = tanhf(am);
            float sp = (av > 20.f) ? av : log1pf(expf(av));
            float p = mu * (sp * (1.f + rvs[r]));
            pred[tid] = p;
            out[((size_t)(b0 + r) * HOR_ + step) * 5 + mm] = p;
        }
        __syncthreads();
    }
}

// ---------------------------------------------------------------------------
// Host orchestration
// ---------------------------------------------------------------------------

extern "C" void kernel_launch(void* const* d_in, const int* in_sizes, int n_in,
                              void* d_out, int out_size, void* d_ws, size_t ws_size,
                              hipStream_t stream) {
    const float* x         = (const float*)d_in[0];
    const int*   sym_id    = (const int*)d_in[1];
    const int*   regime_id = (const int*)d_in[2];
    const float* sym_emb   = (const float*)d_in[3];
    const float* reg_emb   = (const float*)d_in[4];
    const float* in_w      = (const float*)d_in[5];
    const float* in_b      = (const float*)d_in[6];
    const float* enc_qkv_w = (const float*)d_in[7];
    const float* enc_qkv_b = (const float*)d_in[8];
    const float* enc_out_w = (const float*)d_in[9];
    const float* enc_out_b = (const float*)d_in[10];
    const float* ln1_w     = (const float*)d_in[11];
    const float* ln1_b     = (const float*)d_in[12];
    const float* ln2_w     = (const float*)d_in[13];
    const float* ln2_b     = (const float*)d_in[14];
    const float* f1_w      = (const float*)d_in[15];
    const float* f1_b      = (const float*)d_in[16];
    const float* f2_w      = (const float*)d_in[17];
    const float* f2_b      = (const float*)d_in[18];
    const float* pool_w    = (const float*)d_in[19];
    const float* pool_b    = (const float*)d_in[20];
    const float* ctx_w     = (const float*)d_in[21];
    const float* ctx_b     = (const float*)d_in[22];
    const float* gru_wih   = (const float*)d_in[23];
    const float* gru_whh   = (const float*)d_in[24];
    const float* gru_bih   = (const float*)d_in[25];
    const float* gru_bhh   = (const float*)d_in[26];
    const float* mu_w1     = (const float*)d_in[27];
    const float* mu_b1     = (const float*)d_in[28];
    const float* mu_w2     = (const float*)d_in[29];
    const float* mu_b2     = (const float*)d_in[30];
    const float* vol_w1    = (const float*)d_in[31];
    const float* vol_b1    = (const float*)d_in[32];
    const float* vol_w2    = (const float*)d_in[33];
    const float* vol_b2    = (const float*)d_in[34];
    float* out = (float*)d_out;

    const size_t R = (size_t)NB * TT;   // 122880 rows

    size_t cur = 0;
    auto alloc = [&](size_t bytes) -> void* {
        cur = (cur + 255) & ~(size_t)255;
        void* p = (char*)d_ws + cur;
        cur += bytes;
        return p;
    };
    float*  h     = (float*) alloc(R * 128 * 4);       // activations f32
    __bf16* u1    = (__bf16*)alloc(R * 512 * 2);       // union: qkv / ffn1
    __bf16* yln   = (__bf16*)alloc(R * 128 * 2);       // LN output bf16
    __bf16* u2    = (__bf16*)alloc(R * 128 * 2);       // union: a0 / attn_out
    __bf16* w_in  = (__bf16*)alloc((size_t)128 * 64 * 2);
    __bf16* w_qkv = (__bf16*)alloc((size_t)2 * 384 * 128 * 2);
    __bf16* w_out = (__bf16*)alloc((size_t)2 * 128 * 128 * 2);
    __bf16* w_f1  = (__bf16*)alloc((size_t)2 * 512 * 128 * 2);
    __bf16* w_f2  = (__bf16*)alloc((size_t)2 * 128 * 512 * 2);
    __bf16* w_ctx = (__bf16*)alloc((size_t)128 * 160 * 2);
    __bf16* w_gic = (__bf16*)alloc((size_t)384 * 128 * 2);
    __bf16* w_whh = (__bf16*)alloc((size_t)384 * 128 * 2);
    __bf16* w_mu1 = (__bf16*)alloc((size_t)64 * 128 * 2);
    __bf16* w_vol1= (__bf16*)alloc((size_t)64 * 128 * 2);
    __bf16* ctxin = (__bf16*)alloc((size_t)NB * 160 * 2);
    float*  ctxf  = (float*) alloc((size_t)NB * 128 * 4);
    __bf16* ctxb  = (__bf16*)alloc((size_t)NB * 128 * 2);
    float*  gictx = (float*) alloc((size_t)NB * 384 * 4);
    float*  rvbuf = (float*) alloc((size_t)NB * 4);

    auto cvt = [&](const float* src, __bf16* dst, int N, int K, int Kpad, int stride, int soff) {
        int tot = N * Kpad;
        cvt_kernel<<<(tot + 255) / 256, 256, 0, stream>>>(src, dst, N, K, Kpad, stride, soff);
    };
    cvt(in_w,      w_in,   128,  48,  64,  48, 0);
    cvt(enc_qkv_w, w_qkv,  768, 128, 128, 128, 0);
    cvt(enc_out_w, w_out,  256, 128, 128, 128, 0);
    cvt(f1_w,      w_f1,  1024, 128, 128, 128, 0);
    cvt(f2_w,      w_f2,   256, 512, 512, 512, 0);
    cvt(ctx_w,     w_ctx,  128, 152, 160, 152, 0);
    cvt(gru_wih,   w_gic,  384, 128, 128, 133, 5);   // ctx part of W_ih
    cvt(gru_whh,   w_whh,  384, 128, 128, 128, 0);
    cvt(mu_w1,     w_mu1,   64, 128, 128, 128, 0);
    cvt(vol_w1,    w_vol1,  64, 128, 128, 128, 0);

    auto gemm = [&](const __bf16* A, const __bf16* W, const float* bias, const float* resid,
                    float* Cf, __bf16* Cb, int M, int N, int K, int relu) {
        int strips = (M / 16) * (N / 64);
        int blocks = (strips + 7) / 8;
        gemm_bf16_kernel<<<blocks, 256, 0, stream>>>(A, W, bias, resid, Cf, Cb, M, N, K, relu);
    };

    // input projection + positional encoding
    build_in_kernel<<<(int)((R * 64 + 255) / 256), 256, 0, stream>>>(x, sym_id, sym_emb, u2,
                                                                     (int)(R * 64));
    gemm(u2, w_in, in_b, nullptr, h, nullptr, (int)R, 128, 64, 0);
    add_pe_kernel<<<(int)((R * 128 + 255) / 256), 256, 0, stream>>>(h, (int)(R * 128));

    // transformer encoder, pre-LN
    for (int l = 0; l < 2; ++l) {
        ln_kernel<<<(int)(R / 8), 256, 0, stream>>>(h, ln1_w + l * 128, ln1_b + l * 128, yln, (int)R);
        gemm(yln, w_qkv + (size_t)l * 384 * 128, enc_qkv_b + l * 384, nullptr,
             nullptr, u1, (int)R, 384, 128, 0);
        attn_kernel<<<NB * 4, 256, 0, stream>>>(u1, u2);
        gemm(u2, w_out + (size_t)l * 128 * 128, enc_out_b + l * 128, h, h, nullptr,
             (int)R, 128, 128, 0);
        ln_kernel<<<(int)(R / 8), 256, 0, stream>>>(h, ln2_w + l * 128, ln2_b + l * 128, yln, (int)R);
        gemm(yln, w_f1 + (size_t)l * 512 * 128, f1_b + l * 512, nullptr,
             nullptr, u1, (int)R, 512, 128, 1);
        gemm(u1, w_f2 + (size_t)l * 128 * 512, f2_b + l * 128, h, h, nullptr,
             (int)R, 128, 512, 0);
    }

    // pooling, ctx, GRU-invariant input part, realized vol
    pool_kernel<<<NB, 128, 0, stream>>>(h, pool_w, pool_b, sym_id, sym_emb,
                                        regime_id, reg_emb, ctxin);
    gemm(ctxin, w_ctx, ctx_b, nullptr, ctxf, ctxb, NB, 128, 160, 0);
    gemm(ctxb, w_gic, gru_bih, nullptr, gictx, nullptr, NB, 384, 128, 0);
    rv_kernel<<<NB / 256, 256, 0, stream>>>(x, rvbuf, NB);

    // GRU decode
    gru_kernel<<<NB / GM, 256, 0, stream>>>(ctxf, gictx, w_whh, gru_bhh, gru_wih,
                                            w_mu1, mu_b1, mu_w2, mu_b2,
                                            w_vol1, vol_b1, vol_w2, vol_b2,
                                            rvbuf, out);
    (void)in_sizes; (void)n_in; (void)out_size; (void)ws_size;
}
